// SoftMoE_51144470561442
// MI455X (gfx1250) — compile-verified, matrix operand order checked by reading, not run on hardware
//
#include <hip/hip_runtime.h>

typedef __bf16 bf16_t;
typedef __attribute__((ext_vector_type(16))) __bf16 v16bf;
typedef __attribute__((ext_vector_type(8)))  __bf16 v8bf;
typedef __attribute__((ext_vector_type(8)))  float  v8f;
typedef __attribute__((ext_vector_type(2)))  float  v2f;

#define TZ_TOT 16384   // T*Z
#define DDIM   1024
#define NEXP   64
#define PSLOT  32
#define NSLOT  2048    // NEXP*PSLOT
#define HDIM   1024
#define TOK_TILE 16

// ---------------------------------------------------------------------------
// bf16 16x16x32 WMMA fragment load: src is row-major [16 rows, ld elems].
// Lane holds row (lane&15); K chunks [8*(lane>>4), +8) and +16.
// Works for A ([M,K] row-major) and for B supplied as [N,K] row-major.
// ---------------------------------------------------------------------------
__device__ __forceinline__ v16bf load_frag16(const bf16_t* base, int ld) {
  int lane = threadIdx.x & 31;
  const bf16_t* p = base + (size_t)(lane & 15) * (size_t)ld + ((lane >> 4) << 3);
  v8bf lo = *(const v8bf*)(p);
  v8bf hi = *(const v8bf*)(p + 16);
  v16bf r;
#pragma unroll
  for (int i = 0; i < 8; ++i) { r[i] = lo[i]; r[i + 8] = hi[i]; }
  return r;
}

__device__ __forceinline__ v8f wmma_bf16(v16bf a, v16bf b, v8f c) {
  return __builtin_amdgcn_wmma_f32_16x16x32_bf16(false, a, false, b,
                                                 (short)0, c, false, false);
}

// ---------------------------------------------------------------------------
// Kernel 1: elementwise f32 -> bf16 convert (phi_w)
// ---------------------------------------------------------------------------
__global__ __launch_bounds__(256) void cvt_f32_bf16(const float* __restrict__ src,
                                                    bf16_t* __restrict__ dst, int n) {
  int i = blockIdx.x * 256 + threadIdx.x;
  if (i < n) dst[i] = (bf16_t)src[i];
}

// ---------------------------------------------------------------------------
// Kernel 2: tiled transpose + convert: dst[c][r] = (bf16)src[r][c]
// ---------------------------------------------------------------------------
__global__ __launch_bounds__(256) void transpose_cvt(const float* __restrict__ src,
                                                     bf16_t* __restrict__ dst,
                                                     int rows, int cols) {
  __shared__ float tile[32][33];
  int r0 = blockIdx.x * 32, c0 = blockIdx.y * 32;
  int tx = threadIdx.x & 31, ty = threadIdx.x >> 5;   // 32 x 8
#pragma unroll
  for (int i = 0; i < 32; i += 8)
    tile[ty + i][tx] = src[(size_t)(r0 + ty + i) * cols + (c0 + tx)];
  __syncthreads();
#pragma unroll
  for (int i = 0; i < 32; i += 8)
    dst[(size_t)(c0 + ty + i) * rows + (r0 + tx)] = (bf16_t)tile[tx][ty + i];
}

// ---------------------------------------------------------------------------
// Kernel 3: fused routing. One block = 16 tokens.
//  - stage x tile (16x1024) as bf16 in LDS
//  - logits GEMM (bf16 WMMA, 2 N-tiles per A-frag) into 16x2048 f32 LDS tile
//  - dual softmax (over slots -> D, over experts -> C) entirely in LDS
//  - write C [tok, slot] bf16 and D^T [slot, tok] bf16
// Dynamic LDS: 16*1024*2 + 16*2048*4 + 12KB = 176128 B
// ---------------------------------------------------------------------------
__global__ __launch_bounds__(256) void route_kernel(
    const float* __restrict__ x,        // [TZ, D]
    const bf16_t* __restrict__ phiw,    // [NSLOT, D] bf16
    const float* __restrict__ phib,     // [NSLOT]
    bf16_t* __restrict__ Cmat,          // [TZ, NSLOT]
    bf16_t* __restrict__ Dt)            // [NSLOT, TZ]
{
  extern __shared__ char smem[];
  bf16_t* Alds = (bf16_t*)smem;                                   // 32 KB
  float*  Llds = (float*)(smem + TOK_TILE * DDIM * 2);            // 128 KB
  float*  stats = (float*)(smem + TOK_TILE * DDIM * 2 + TOK_TILE * NSLOT * 4);
  float* maxD = stats;                 // [16][64]
  float* sumD = maxD + 16 * 64;        // [16][64]
  float* maxC = sumD + 16 * 64;        // [16][32]
  float* sumC = maxC + 16 * 32;        // [16][32]

  const int tok0 = blockIdx.x * TOK_TILE;
  const int tid  = threadIdx.x;
  const int wave = tid >> 5;
  const int lane = tid & 31;

  // stage + convert x tile (coalesced along D)
  for (int i = tid; i < TOK_TILE * DDIM; i += 256) {
    int m = i >> 10, k = i & 1023;
    Alds[i] = (bf16_t)x[(size_t)(tok0 + m) * DDIM + k];
  }
  __syncthreads();

  // logits GEMM: each wave sweeps N-tile PAIRS (A-frag reused for 2 WMMAs)
  for (int ntp = wave; ntp < NSLOT / 32; ntp += 8) {
    int nt0 = ntp * 2;
    v8f acc0 = {}, acc1 = {};
    const bf16_t* Bb0 = phiw + (size_t)nt0 * 16 * DDIM;
    const bf16_t* Bb1 = Bb0 + 16 * DDIM;
#pragma unroll 4
    for (int kk = 0; kk < DDIM; kk += 32) {
      v16bf a  = load_frag16(Alds + kk, DDIM);
      v16bf b0 = load_frag16(Bb0 + kk, DDIM);
      v16bf b1 = load_frag16(Bb1 + kk, DDIM);
      acc0 = wmma_bf16(a, b0, acc0);
      acc1 = wmma_bf16(a, b1, acc1);
    }
    int rbase = (lane >> 4) * 8;
    int n0 = nt0 * 16 + (lane & 15);
    float bias0 = phib[n0];
    float bias1 = phib[n0 + 16];
#pragma unroll
    for (int r = 0; r < 8; ++r) {
      Llds[(rbase + r) * NSLOT + n0]      = acc0[r] + bias0;
      Llds[(rbase + r) * NSLOT + n0 + 16] = acc1[r] + bias1;
    }
  }
  __syncthreads();

  // dispatch stats: softmax over the 32 slots of each (token m, expert e)
  for (int g = tid; g < 16 * 64; g += 256) {
    int m = g >> 6, e = g & 63;
    const float* row = Llds + m * NSLOT + e * 32;
    float mx = row[0];
#pragma unroll
    for (int p = 1; p < 32; ++p) mx = fmaxf(mx, row[p]);
    float s = 0.f;
#pragma unroll
    for (int p = 0; p < 32; ++p) s += __expf(row[p] - mx);
    maxD[g] = mx; sumD[g] = s;
  }
  // combine stats: softmax over the 64 experts of each (token m, slot p)
  for (int g = tid; g < 16 * 32; g += 256) {
    int m = g >> 5, p = g & 31;
    const float* col = Llds + m * NSLOT + p;
    float mx = col[0];
#pragma unroll
    for (int e = 1; e < 64; ++e) mx = fmaxf(mx, col[e * 32]);
    float s = 0.f;
#pragma unroll
    for (int e = 0; e < 64; ++e) s += __expf(col[e * 32] - mx);
    maxC[g] = mx; sumC[g] = s;
  }
  __syncthreads();

  // write C row-major (coalesced along slot index)
  for (int i = tid; i < 16 * NSLOT; i += 256) {
    int m = i >> 11, j = i & (NSLOT - 1);
    int p = j & 31;
    float v = Llds[m * NSLOT + j];
    float c = __expf(v - maxC[m * 32 + p]) / sumC[m * 32 + p];
    Cmat[(size_t)(tok0 + m) * NSLOT + j] = (bf16_t)c;
  }
  // write D^T [slot, token]: consecutive threads -> consecutive tokens (32B runs)
  for (int i = tid; i < 16 * NSLOT; i += 256) {
    int j = i >> 4, m = i & 15;
    int e = j >> 5;
    float v = Llds[m * NSLOT + j];
    float d = __expf(v - maxD[m * 64 + e]) / sumD[m * 64 + e];
    Dt[(size_t)j * TZ_TOT + tok0 + m] = (bf16_t)d;
  }
}

// ---------------------------------------------------------------------------
// Kernel 4: X_tilde = D^T @ x  -> [NSLOT, DDIM] f32. K = TZ.
// A = Dt [NSLOT, TZ] bf16 ; B = xT [DDIM, TZ] bf16 (both K-contiguous).
// Register-blocked: each wave owns 2 M-tiles x 4 N-tiles (8 accumulators).
// grid (NSLOT/32 = 64, 2); (blockIdx.y*8 + wave) selects the N-quad.
// ---------------------------------------------------------------------------
__global__ __launch_bounds__(256) void gemm_xtilde(const bf16_t* __restrict__ Dt,
                                                   const bf16_t* __restrict__ xT,
                                                   float* __restrict__ Xt) {
  int wave = threadIdx.x >> 5, lane = threadIdx.x & 31;
  int mt0 = blockIdx.x * 2;
  int nt0 = (blockIdx.y * 8 + wave) * 4;       // 16 wave-slots * 4 = 64 N-tiles
  const bf16_t* A0 = Dt + (size_t)mt0 * 16 * TZ_TOT;
  const bf16_t* A1 = A0 + (size_t)16 * TZ_TOT;
  const bf16_t* B0 = xT + (size_t)nt0 * 16 * TZ_TOT;
  v8f acc[2][4] = {};
  for (int kk = 0; kk < TZ_TOT; kk += 32) {
    v16bf a0 = load_frag16(A0 + kk, TZ_TOT);
    v16bf a1 = load_frag16(A1 + kk, TZ_TOT);
#pragma unroll
    for (int j = 0; j < 4; ++j) {
      v16bf b = load_frag16(B0 + (size_t)j * 16 * TZ_TOT + kk, TZ_TOT);
      acc[0][j] = wmma_bf16(a0, b, acc[0][j]);
      acc[1][j] = wmma_bf16(a1, b, acc[1][j]);
    }
  }
  int rbase = (lane >> 4) * 8;
#pragma unroll
  for (int mi = 0; mi < 2; ++mi)
#pragma unroll
    for (int j = 0; j < 4; ++j) {
      int n = (nt0 + j) * 16 + (lane & 15);
#pragma unroll
      for (int r = 0; r < 8; ++r)
        Xt[(size_t)((mt0 + mi) * 16 + rbase + r) * DDIM + n] = acc[mi][j][r];
    }
}

// ---------------------------------------------------------------------------
// Kernel 5: clip + LayerNorm(no affine) + tanh per slot row. block = row.
// ---------------------------------------------------------------------------
__global__ __launch_bounds__(256) void ln_tanh_kernel(const float* __restrict__ Xt,
                                                      float* __restrict__ Xn) {
  __shared__ float sbuf[256];
  int row = blockIdx.x, tid = threadIdx.x;
  const float* src = Xt + (size_t)row * DDIM;
  float v[4]; float s = 0.f;
#pragma unroll
  for (int i = 0; i < 4; ++i) {
    float t = src[tid + 256 * i];
    t = fminf(fmaxf(t, -33000.f), 65000.f);
    v[i] = t; s += t;
  }
  sbuf[tid] = s; __syncthreads();
  for (int off = 128; off > 0; off >>= 1) {
    if (tid < off) sbuf[tid] += sbuf[tid + off];
    __syncthreads();
  }
  float mu = sbuf[0] * (1.f / DDIM);
  __syncthreads();
  float q = 0.f;
#pragma unroll
  for (int i = 0; i < 4; ++i) { float d = v[i] - mu; q += d * d; }
  sbuf[tid] = q; __syncthreads();
  for (int off = 128; off > 0; off >>= 1) {
    if (tid < off) sbuf[tid] += sbuf[tid + off];
    __syncthreads();
  }
  float inv = rsqrtf(sbuf[0] * (1.f / DDIM) + 1e-5f);
#pragma unroll
  for (int i = 0; i < 4; ++i)
    Xn[(size_t)row * DDIM + tid + 256 * i] = tanhf((v[i] - mu) * inv);
}

// ---------------------------------------------------------------------------
// Kernel 6: per-expert GEMM in f32 WMMA 16x16x4 (streams expert_w once;
// pure bandwidth-bound, so no conversion pass). Y_tilde written TRANSPOSED
// bf16 [H, slot] (+bias) to be the K-contiguous B operand of the combine GEMM.
// ---------------------------------------------------------------------------
__global__ __launch_bounds__(256) void expert_gemm(const float* __restrict__ Xn,
                                                   const float* __restrict__ Ew,
                                                   const float* __restrict__ Eb,
                                                   bf16_t* __restrict__ YtT) {
  int n = blockIdx.x;
  int wave = threadIdx.x >> 5, lane = threadIdx.x & 31;
  const float* A0 = Xn + (size_t)n * PSLOT * DDIM;
  const float* B0 = Ew + (size_t)n * DDIM * HDIM;
  int koff = (lane >> 4) * 2;
  for (int t = wave; t < 128; t += 8) {
    int mt = t & 1, ht = t >> 1;
    int arow = mt * 16 + (lane & 15);
    int h    = ht * 16 + (lane & 15);
    const float* Arow = A0 + (size_t)arow * DDIM + koff;
    v8f acc = {};
    for (int k = 0; k < DDIM; k += 4) {
      v2f a = *(const v2f*)(Arow + k);
      v2f b;
      b.x = B0[(size_t)(k + koff) * HDIM + h];
      b.y = B0[(size_t)(k + koff + 1) * HDIM + h];
      acc = __builtin_amdgcn_wmma_f32_16x16x4_f32(false, a, false, b,
                                                  (short)0, acc, false, false);
    }
    float bias = Eb[(size_t)n * HDIM + h];
    int rbase = (lane >> 4) * 8;
    v8bf pack;
#pragma unroll
    for (int r = 0; r < 8; ++r) pack[r] = (bf16_t)(acc[r] + bias);
    *(v8bf*)(YtT + (size_t)h * NSLOT + n * PSLOT + mt * 16 + rbase) = pack;
  }
}

// ---------------------------------------------------------------------------
// Kernel 7: Y = C @ Y_tilde -> [TZ, H] f32. K = NSLOT.
// A = Cmat [TZ, NSLOT] bf16 ; B = YtT [H, NSLOT] bf16.
// Register-blocked 2 M-tiles x 4 N-tiles per wave. grid (TZ/32 = 512, 2).
// ---------------------------------------------------------------------------
__global__ __launch_bounds__(256) void gemm_out(const bf16_t* __restrict__ Cmat,
                                                const bf16_t* __restrict__ YtT,
                                                float* __restrict__ Y) {
  int wave = threadIdx.x >> 5, lane = threadIdx.x & 31;
  int mt0 = blockIdx.x * 2;
  int nt0 = (blockIdx.y * 8 + wave) * 4;       // 16 wave-slots * 4 = 64 H-tiles
  const bf16_t* A0 = Cmat + (size_t)mt0 * 16 * NSLOT;
  const bf16_t* A1 = A0 + (size_t)16 * NSLOT;
  const bf16_t* B0 = YtT + (size_t)nt0 * 16 * NSLOT;
  v8f acc[2][4] = {};
  for (int kk = 0; kk < NSLOT; kk += 32) {
    v16bf a0 = load_frag16(A0 + kk, NSLOT);
    v16bf a1 = load_frag16(A1 + kk, NSLOT);
#pragma unroll
    for (int j = 0; j < 4; ++j) {
      v16bf b = load_frag16(B0 + (size_t)j * 16 * NSLOT + kk, NSLOT);
      acc[0][j] = wmma_bf16(a0, b, acc[0][j]);
      acc[1][j] = wmma_bf16(a1, b, acc[1][j]);
    }
  }
  int rbase = (lane >> 4) * 8;
#pragma unroll
  for (int mi = 0; mi < 2; ++mi)
#pragma unroll
    for (int j = 0; j < 4; ++j) {
      int h = (nt0 + j) * 16 + (lane & 15);
#pragma unroll
      for (int r = 0; r < 8; ++r)
        Y[(size_t)((mt0 + mi) * 16 + rbase + r) * HDIM + h] = acc[mi][j][r];
    }
}

// ---------------------------------------------------------------------------
extern "C" void kernel_launch(void* const* d_in, const int* in_sizes, int n_in,
                              void* d_out, int out_size, void* d_ws, size_t ws_size,
                              hipStream_t stream) {
  (void)in_sizes; (void)n_in; (void)out_size; (void)ws_size;
  const float* x    = (const float*)d_in[0];  // [TZ, D]
  const float* phiw = (const float*)d_in[1];  // [NSLOT, D]
  const float* phib = (const float*)d_in[2];  // [NSLOT]
  const float* ew   = (const float*)d_in[3];  // [NEXP, D, H]
  const float* eb   = (const float*)d_in[4];  // [NEXP, H]
  float* Y = (float*)d_out;                   // [TZ, H]

  char* w = (char*)d_ws;
  bf16_t* phiw_bf = (bf16_t*)w; w += (size_t)NSLOT * DDIM * 2;   //   4 MB
  bf16_t* xT      = (bf16_t*)w; w += (size_t)DDIM * TZ_TOT * 2;  //  33 MB
  bf16_t* Cmat    = (bf16_t*)w; w += (size_t)TZ_TOT * NSLOT * 2; //  67 MB
  bf16_t* Dt      = (bf16_t*)w; w += (size_t)NSLOT * TZ_TOT * 2; //  67 MB
  float*  Xt      = (float*)w;  w += (size_t)NSLOT * DDIM * 4;   // 8.4 MB
  float*  Xn      = (float*)w;  w += (size_t)NSLOT * DDIM * 4;   // 8.4 MB
  bf16_t* YtT     = (bf16_t*)w; w += (size_t)HDIM * NSLOT * 2;   // 4.2 MB

  // 1) phi_w -> bf16
  int npw = NSLOT * DDIM;
  cvt_f32_bf16<<<(npw + 255) / 256, 256, 0, stream>>>(phiw, phiw_bf, npw);
  // 2) x -> x^T bf16
  dim3 tgrid(TZ_TOT / 32, DDIM / 32);
  transpose_cvt<<<tgrid, 256, 0, stream>>>(x, xT, TZ_TOT, DDIM);
  // 3) fused routing (logits GEMM + dual softmax)
  size_t route_lds = (size_t)TOK_TILE * DDIM * 2 + (size_t)TOK_TILE * NSLOT * 4
                   + (16 * 64 * 2 + 16 * 32 * 2) * sizeof(float);   // 176128 B
  (void)hipFuncSetAttribute((const void*)route_kernel,
                            hipFuncAttributeMaxDynamicSharedMemorySize,
                            (int)route_lds);
  route_kernel<<<TZ_TOT / TOK_TILE, 256, route_lds, stream>>>(x, phiw_bf, phib,
                                                              Cmat, Dt);
  // 4) X_tilde = D^T @ x  (register-blocked bf16 WMMA)
  gemm_xtilde<<<dim3(NSLOT / 32, 2), 256, 0, stream>>>(Dt, xT, Xt);
  // 5) clip + LN + tanh
  ln_tanh_kernel<<<NSLOT, 256, 0, stream>>>(Xt, Xn);
  // 6) per-expert GEMM (f32 WMMA), writes Y_tilde^T bf16 + bias
  expert_gemm<<<NEXP, 256, 0, stream>>>(Xn, ew, eb, YtT);
  // 7) Y = C @ Y_tilde  (register-blocked bf16 WMMA)
  gemm_out<<<dim3(TZ_TOT / 32, 2), 256, 0, stream>>>(Cmat, YtT, Y);
}
